// TemporalFusionDeformMiddle_68255620268147
// MI455X (gfx1250) — compile-verified, matrix operand order checked by reading, not run on hardware
//
#include <hip/hip_runtime.h>
#include <hip/hip_bf16.h>
#include <math.h>

// ---- problem constants (from reference) ----
#define HEADS   4
#define SAMPLES 8
#define CB      64          // channels
#define DD      16
#define HH      64
#define WW      64
#define HD      16          // head dim
#define NS      65536       // D*H*W spatial
#define BN_EPS  1e-5f

typedef __attribute__((ext_vector_type(16))) __bf16 v16bf;
typedef __attribute__((ext_vector_type(8)))  __bf16 v8bf;
typedef __attribute__((ext_vector_type(8)))  float  v8f;

union FragU { v16bf v; v8bf h[2]; };

// ---------------- layout-transpose + fp32->bf16 via async-to-LDS ----------------
// src: [ci][NS] fp32   ->   dst: [n][64] bf16 (channel-contiguous for WMMA B frags)
__global__ __launch_bounds__(256)
void transpose_cvt_kernel(const float* __restrict__ src, __bf16* __restrict__ dst) {
    __shared__ float tile[64][65];
    const int tid = threadIdx.x;
    const int n0  = blockIdx.x * 64;
    // async global -> LDS (raw 4B per lane), coalesced along n
    for (int i = tid; i < 64 * 64; i += 256) {
        int ci = i >> 6, dn = i & 63;
        unsigned lds = (unsigned)(size_t)&tile[ci][dn];
        unsigned long long ga =
            (unsigned long long)(const void*)(src + (size_t)ci * NS + n0 + dn);
        asm volatile("global_load_async_to_lds_b32 %0, %1, off" :: "v"(lds), "v"(ga)
                     : "memory");
    }
    asm volatile("s_wait_asynccnt 0x0" ::: "memory");
    __syncthreads();
    // LDS -> bf16, coalesced along ci
    for (int i = tid; i < 64 * 64; i += 256) {
        int dn = i >> 6, ci = i & 63;
        dst[(size_t)(n0 + dn) * CB + ci] = (__bf16)tile[ci][dn];
    }
}

__global__ void copy_f32_kernel(const float* __restrict__ src, float* __restrict__ dst, int n) {
    int i = blockIdx.x * blockDim.x + threadIdx.x;
    for (; i < n; i += gridDim.x * blockDim.x) dst[i] = src[i];
}

__global__ void cvt_bf16_kernel(const float* __restrict__ src, __bf16* __restrict__ dst, int n) {
    int i = blockIdx.x * blockDim.x + threadIdx.x;
    for (; i < n; i += gridDim.x * blockDim.x) dst[i] = (__bf16)src[i];
}

// conv weights (co,ci,tap) fp32 -> (tap,co,ci) bf16  (A-fragment = 2 contiguous 16B runs)
__global__ void cvt_w_conv_kernel(const float* __restrict__ W, __bf16* __restrict__ Wb,
                                  int Cout) {
    int i = blockIdx.x * blockDim.x + threadIdx.x;
    int total = Cout * CB * 27;
    if (i >= total) return;
    int tap = i % 27, rem = i / 27;
    int ci = rem % CB, co = rem / CB;
    Wb[((size_t)tap * Cout + co) * CB + ci] = (__bf16)W[i];
}

// load one 16x32 bf16 fragment from global: two contiguous 16B runs per lane
__device__ __forceinline__ v16bf load_frag(const __bf16* base) {
    FragU u;
    u.h[0] = *(const v8bf*)(base);
    u.h[1] = *(const v8bf*)(base + 16);
    return u.v;
}

// ---------------- 1x1x1 conv as WMMA GEMM ----------------
// Wb bf16 [Cout][64], X bf16 [n][64].
// MODE 0: store fp32 transposed [n][Cout] (feeds gather stage)
// MODE 1: +bias+BN+ReLU, store fp32 [co][NS] into concat output
template<int MODE>
__global__ __launch_bounds__(256)
void gemm1x1_kernel(const __bf16* __restrict__ Wb, const float* __restrict__ bias,
                    const __bf16* __restrict__ X, float* __restrict__ Out, int Cout,
                    const float* __restrict__ g, const float* __restrict__ bt,
                    const float* __restrict__ mu, const float* __restrict__ var) {
    const int lane  = threadIdx.x & 31;
    const int wave  = threadIdx.x >> 5;
    const int lhalf = lane >> 4;
    const int lr    = lane & 15;
    const int m0    = wave * 16;
    if (m0 >= Cout) return;                 // wave-uniform
    const int n  = blockIdx.x * 16 + lr;    // this lane's column

    v8f acc = {};
#pragma unroll
    for (int k0 = 0; k0 < CB; k0 += 32) {
        v16bf a = load_frag(Wb + (size_t)(m0 + lr) * CB + k0 + 8 * lhalf);
        v16bf b = load_frag(X + (size_t)n * CB + k0 + 8 * lhalf);
        acc = __builtin_amdgcn_wmma_f32_16x16x32_bf16(false, a, false, b, (short)0, acc,
                                                      false, false);
    }
    if (MODE == 0) {  // transposed store: contiguous run of 8 floats per lane
        float4 s0, s1;
        s0.x = acc[0] + bias[m0 + 8 * lhalf + 0];
        s0.y = acc[1] + bias[m0 + 8 * lhalf + 1];
        s0.z = acc[2] + bias[m0 + 8 * lhalf + 2];
        s0.w = acc[3] + bias[m0 + 8 * lhalf + 3];
        s1.x = acc[4] + bias[m0 + 8 * lhalf + 4];
        s1.y = acc[5] + bias[m0 + 8 * lhalf + 5];
        s1.z = acc[6] + bias[m0 + 8 * lhalf + 6];
        s1.w = acc[7] + bias[m0 + 8 * lhalf + 7];
        float* o = Out + (size_t)n * Cout + m0 + 8 * lhalf;
        *(float4*)(o)     = s0;
        *(float4*)(o + 4) = s1;
    } else {
#pragma unroll
        for (int r = 0; r < 8; ++r) {
            int co = m0 + r + 8 * lhalf;
            float v = acc[r] + bias[co];
            float s = g[co] * rsqrtf(var[co] + BN_EPS);
            v = (v - mu[co]) * s + bt[co];
            Out[(size_t)co * NS + n] = fmaxf(v, 0.0f);
        }
    }
}

// ---------------- 3x3x3 conv (pad=1) as implicit-GEMM WMMA ----------------
// Wb bf16 [tap][Cout][64], X bf16 [n][64].
// The 3x3x18-point halo patch (64 ch each) is staged ONCE per block into LDS via
// async-to-LDS b128 (OOB points pre-zeroed), then every wave builds B fragments
// with 16B ds loads. 27 taps x 2 chunks = 54 WMMA / 16x16 tile.
// ACT 0: +bias, exact GELU -> bf16 [n][64].   ACT 1: tanh -> fp32 [n][96].
#define PTS (3 * 3 * 18)      // 162 halo points
template<int ACT>
__global__ __launch_bounds__(256)
void conv3x3x3_kernel(const __bf16* __restrict__ Wb, const float* __restrict__ bias,
                      const __bf16* __restrict__ X,
                      __bf16* __restrict__ out_bf, float* __restrict__ out_f, int Cout) {
    __shared__ __bf16 tileX[PTS * CB];      // 162 * 64 * 2B = 20.25 KB

    const int lane  = threadIdx.x & 31;
    const int wave  = threadIdx.x >> 5;
    const int lhalf = lane >> 4;
    const int lr    = lane & 15;
    const int m0    = wave * 16;
    const int n0 = blockIdx.x * 16;
    const int z0 = n0 >> 12;
    const int y0 = (n0 >> 6) & 63;
    const int x0 = n0 & 63;            // tile of 16 contiguous x, never crosses a row

    // ---- stage input patch: 162 points x 64ch, 16B (8 ch) per task ----
    for (int i = threadIdx.x; i < PTS * 8; i += blockDim.x) {
        const int p = i >> 3, seg = i & 7;
        const int zz = p / 54, yy = (p / 18) % 3, xx = p % 18;
        const int z = z0 - 1 + zz, y = y0 - 1 + yy, x = x0 - 1 + xx;
        const bool ok = (z >= 0) & (z < DD) & (y >= 0) & (y < HH) & (x >= 0) & (x < WW);
        __bf16* lp = &tileX[(size_t)p * CB + seg * 8];
        if (ok) {
            const size_t sp = ((size_t)z * HH + y) * WW + x;
            unsigned lds = (unsigned)(size_t)lp;
            unsigned long long ga =
                (unsigned long long)(const void*)(X + sp * CB + seg * 8);
            asm volatile("global_load_async_to_lds_b128 %0, %1, off" :: "v"(lds), "v"(ga)
                         : "memory");
        } else {
            *(v8bf*)lp = (v8bf){};     // zero halo (ds store)
        }
    }
    asm volatile("s_wait_asynccnt 0x0" ::: "memory");
    __syncthreads();

    if (m0 >= Cout) return;            // wave-uniform

    v8f acc = {};
    for (int tap = 0; tap < 27; ++tap) {
        const int dz = tap / 9, dy = (tap / 3) % 3, dx = tap % 3;   // 0..2 (halo space)
        const __bf16* wbase = Wb + ((size_t)tap * Cout + (m0 + lr)) * CB;
        const __bf16* xbase = &tileX[(size_t)((dz * 3 + dy) * 18 + (lr + dx)) * CB];
#pragma unroll
        for (int k0 = 0; k0 < CB; k0 += 32) {
            v16bf a = load_frag(wbase + k0 + 8 * lhalf);   // global (L0-resident weights)
            FragU ub;                                      // LDS 16B reads
            ub.h[0] = *(const v8bf*)(xbase + k0 + 8 * lhalf);
            ub.h[1] = *(const v8bf*)(xbase + k0 + 16 + 8 * lhalf);
            acc = __builtin_amdgcn_wmma_f32_16x16x32_bf16(false, a, false, ub.v, (short)0,
                                                          acc, false, false);
        }
    }
    const int n = n0 + lr;
    if (ACT == 0) {  // GELU -> bf16 [n][64], one 16B store per lane
        v8bf ov;
#pragma unroll
        for (int r = 0; r < 8; ++r) {
            float v = acc[r] + bias[m0 + 8 * lhalf + r];
            v = 0.5f * v * (1.0f + erff(v * 0.70710678118654752f));
            ov[r] = (__bf16)v;
        }
        *(v8bf*)(out_bf + (size_t)n * CB + m0 + 8 * lhalf) = ov;
    } else {         // tanh -> fp32 [n][96], two float4 stores per lane
        float4 s0, s1;
        s0.x = tanhf(acc[0]); s0.y = tanhf(acc[1]); s0.z = tanhf(acc[2]); s0.w = tanhf(acc[3]);
        s1.x = tanhf(acc[4]); s1.y = tanhf(acc[5]); s1.z = tanhf(acc[6]); s1.w = tanhf(acc[7]);
        float* o = out_f + (size_t)n * 96 + m0 + 8 * lhalf;
        *(float4*)(o)     = s0;
        *(float4*)(o + 4) = s1;
    }
}

// ---------------- deformable sampling + softmax attention ----------------
// kv fp32 [voxel][128], q fp32 [voxel][64], off fp32 [voxel][96] -> vector gathers.
// fused bf16 [voxel][64] (feeds final WMMA GEMM).
__global__ __launch_bounds__(256)
void sample_attn_kernel(const float* __restrict__ kv, const float* __restrict__ q,
                        const float* __restrict__ off, __bf16* __restrict__ fused) {
    const int gid = blockIdx.x * blockDim.x + threadIdx.x;
    if (gid >= HEADS * NS) return;
    const int m = gid >> 16;
    const int n = gid & (NS - 1);
    const int z = n >> 12, y = (n >> 6) & 63, x = n & 63;

    // base grid: comp0 uses zz -> ix/W; comp1 uses yy -> iy/H; comp2 uses xx -> iz/D
    const float bz = -1.0f + 2.0f * (float)z / (float)(DD - 1);
    const float by = -1.0f + 2.0f * (float)y / (float)(HH - 1);
    const float bx = -1.0f + 2.0f * (float)x / (float)(WW - 1);

    float4 qv[4];
    {
        const float4* qp = (const float4*)(q + (size_t)n * CB + m * HD);
#pragma unroll
        for (int c4 = 0; c4 < 4; ++c4) {
            float4 t = qp[c4];
            qv[c4].x = t.x * 0.25f; qv[c4].y = t.y * 0.25f;   // HD^-0.5
            qv[c4].z = t.z * 0.25f; qv[c4].w = t.w * 0.25f;
        }
    }

    float ixs[SAMPLES], iys[SAMPLES], izs[SAMPLES], logit[SAMPLES];
    const float* offn = off + (size_t)n * 96 + m * SAMPLES * 3;

#pragma unroll
    for (int s = 0; s < SAMPLES; ++s) {
        float o0 = offn[s * 3 + 0], o1 = offn[s * 3 + 1], o2 = offn[s * 3 + 2];
        float g0 = fminf(fmaxf(bz + o0 / (float)DD, -1.0f), 1.0f);
        float g1 = fminf(fmaxf(by + o1 / (float)HH, -1.0f), 1.0f);
        float g2 = fminf(fmaxf(bx + o2 / (float)WW, -1.0f), 1.0f);
        float ix = (g0 + 1.0f) * 0.5f * (WW - 1);
        float iy = (g1 + 1.0f) * 0.5f * (HH - 1);
        float iz = (g2 + 1.0f) * 0.5f * (DD - 1);
        ixs[s] = ix; iys[s] = iy; izs[s] = iz;

        float fx = floorf(ix), fy = floorf(iy), fz = floorf(iz);
        float wx = ix - fx, wy = iy - fy, wz = iz - fz;
        int x0i = min(max((int)fx, 0), WW - 1), x1i = min(x0i + 1, WW - 1);
        int y0i = min(max((int)fy, 0), HH - 1), y1i = min(y0i + 1, HH - 1);
        int z0i = min(max((int)fz, 0), DD - 1), z1i = min(z0i + 1, DD - 1);

        float dot = 0.0f;
#pragma unroll
        for (int corner = 0; corner < 8; ++corner) {
            int cz = corner >> 2, cy = (corner >> 1) & 1, cx = corner & 1;
            float wgt = (cz ? wz : 1.0f - wz) * (cy ? wy : 1.0f - wy) * (cx ? wx : 1.0f - wx);
            size_t sp = ((size_t)(cz ? z1i : z0i) * HH + (cy ? y1i : y0i)) * WW +
                        (cx ? x1i : x0i);
            const float4* kr = (const float4*)(kv + sp * 128 + m * 2 * HD);  // key run
            float a = 0.0f;
#pragma unroll
            for (int c4 = 0; c4 < 4; ++c4) {
                float4 kx = kr[c4];
                a += qv[c4].x * kx.x + qv[c4].y * kx.y + qv[c4].z * kx.z + qv[c4].w * kx.w;
            }
            dot += wgt * a;
        }
        logit[s] = dot;
    }

    // softmax over samples
    float mx = logit[0];
#pragma unroll
    for (int s = 1; s < SAMPLES; ++s) mx = fmaxf(mx, logit[s]);
    float ssum = 0.0f;
#pragma unroll
    for (int s = 0; s < SAMPLES; ++s) { logit[s] = expf(logit[s] - mx); ssum += logit[s]; }
    const float inv = 1.0f / ssum;

    float4 fo[4];
#pragma unroll
    for (int c4 = 0; c4 < 4; ++c4) fo[c4] = make_float4(0.f, 0.f, 0.f, 0.f);

#pragma unroll
    for (int s = 0; s < SAMPLES; ++s) {
        float attn = logit[s] * inv;
        float ix = ixs[s], iy = iys[s], iz = izs[s];
        float fx = floorf(ix), fy = floorf(iy), fz = floorf(iz);
        float wx = ix - fx, wy = iy - fy, wz = iz - fz;
        int x0i = min(max((int)fx, 0), WW - 1), x1i = min(x0i + 1, WW - 1);
        int y0i = min(max((int)fy, 0), HH - 1), y1i = min(y0i + 1, HH - 1);
        int z0i = min(max((int)fz, 0), DD - 1), z1i = min(z0i + 1, DD - 1);
#pragma unroll
        for (int corner = 0; corner < 8; ++corner) {
            int cz = corner >> 2, cy = (corner >> 1) & 1, cx = corner & 1;
            float coef = attn * (cz ? wz : 1.0f - wz) * (cy ? wy : 1.0f - wy) *
                         (cx ? wx : 1.0f - wx);
            size_t sp = ((size_t)(cz ? z1i : z0i) * HH + (cy ? y1i : y0i)) * WW +
                        (cx ? x1i : x0i);
            const float4* vr = (const float4*)(kv + sp * 128 + m * 2 * HD + HD);  // value run
#pragma unroll
            for (int c4 = 0; c4 < 4; ++c4) {
                float4 vx = vr[c4];
                fo[c4].x += coef * vx.x; fo[c4].y += coef * vx.y;
                fo[c4].z += coef * vx.z; fo[c4].w += coef * vx.w;
            }
        }
    }
    // store 16 contiguous bf16 (32B) into [n][64]
    v8bf o0, o1;
#pragma unroll
    for (int c4 = 0; c4 < 2; ++c4) {
        o0[c4 * 4 + 0] = (__bf16)fo[c4].x; o0[c4 * 4 + 1] = (__bf16)fo[c4].y;
        o0[c4 * 4 + 2] = (__bf16)fo[c4].z; o0[c4 * 4 + 3] = (__bf16)fo[c4].w;
        o1[c4 * 4 + 0] = (__bf16)fo[c4 + 2].x; o1[c4 * 4 + 1] = (__bf16)fo[c4 + 2].y;
        o1[c4 * 4 + 2] = (__bf16)fo[c4 + 2].z; o1[c4 * 4 + 3] = (__bf16)fo[c4 + 2].w;
    }
    __bf16* fp = fused + (size_t)n * CB + m * HD;
    *(v8bf*)(fp)     = o0;
    *(v8bf*)(fp + 8) = o1;
}

// ---------------- host orchestration ----------------
extern "C" void kernel_launch(void* const* d_in, const int* in_sizes, int n_in,
                              void* d_out, int out_size, void* d_ws, size_t ws_size,
                              hipStream_t stream) {
    const float* mid    = (const float*)d_in[0];
    const float* occ    = (const float*)d_in[1];
    const float* prev   = (const float*)d_in[2];
    const float* off1_w = (const float*)d_in[3];
    const float* off1_b = (const float*)d_in[4];
    const float* off2_w = (const float*)d_in[5];
    const float* kv_w   = (const float*)d_in[6];
    const float* kv_b   = (const float*)d_in[7];
    const float* q_w    = (const float*)d_in[8];
    const float* q_b    = (const float*)d_in[9];
    const float* out_w  = (const float*)d_in[10];
    const float* out_b  = (const float*)d_in[11];
    const float* bn_g   = (const float*)d_in[12];
    const float* bn_bt  = (const float*)d_in[13];
    const float* bn_mu  = (const float*)d_in[14];
    const float* bn_v   = (const float*)d_in[15];
    float* out = (float*)d_out;

    // workspace layout
    char* w = (char*)d_ws;
    __bf16* bf_mid = (__bf16*)w;  w += (size_t)NS * CB * 2;          // [n][64]   8 MB
    __bf16* bf_cur = (__bf16*)w;  w += (size_t)NS * CB * 2;          //           8 MB
    float*  kvf    = (float*)w;   w += (size_t)NS * 2 * CB * 4;      // [n][128] 32 MB
    float*  qf     = (float*)w;   w += (size_t)NS * CB * 4;          // [n][64]  16 MB
    __bf16* o1     = (__bf16*)w;  w += (size_t)NS * CB * 2;          // [n][64]   8 MB
    float*  offf   = (float*)w;   w += (size_t)NS * 96 * 4;          // [n][96]  24 MB
    __bf16* fusedb = (__bf16*)w;  w += (size_t)NS * CB * 2;          // [n][64]   8 MB
    __bf16* wb_kv  = (__bf16*)w;  w += (size_t)128 * CB * 2;
    __bf16* wb_q   = (__bf16*)w;  w += (size_t)CB * CB * 2;
    __bf16* wb_out = (__bf16*)w;  w += (size_t)CB * CB * 2;
    __bf16* wb_o1  = (__bf16*)w;  w += (size_t)27 * CB * CB * 2;     // [tap][64][64]
    __bf16* wb_o2  = (__bf16*)w;  w += (size_t)27 * 96 * CB * 2;     // [tap][96][64]
    (void)ws_size; (void)n_in; (void)in_sizes; (void)out_size;

    const int nEl = CB * NS;
    const int tilesN = NS / 16;                // 4096

    // weight prep (bf16, fragment-friendly layouts)
    cvt_bf16_kernel<<<32, 256, 0, stream>>>(kv_w, wb_kv, 128 * CB);
    cvt_bf16_kernel<<<16, 256, 0, stream>>>(q_w, wb_q, CB * CB);
    cvt_bf16_kernel<<<16, 256, 0, stream>>>(out_w, wb_out, CB * CB);
    cvt_w_conv_kernel<<<(64 * CB * 27 + 255) / 256, 256, 0, stream>>>(off1_w, wb_o1, 64);
    cvt_w_conv_kernel<<<(96 * CB * 27 + 255) / 256, 256, 0, stream>>>(off2_w, wb_o2, 96);

    transpose_cvt_kernel<<<NS / 64, 256, 0, stream>>>(mid, bf_mid);
    copy_f32_kernel<<<4096, 256, 0, stream>>>(mid, out, nEl);        // concat slot 0

    // kv depends only on mid -> compute once for both passes
    gemm1x1_kernel<0><<<tilesN, 256, 0, stream>>>(wb_kv, kv_b, bf_mid, kvf, 128,
                                                  nullptr, nullptr, nullptr, nullptr);

    for (int pass = 0; pass < 2; ++pass) {
        const float* cur = pass ? prev : occ;
        transpose_cvt_kernel<<<NS / 64, 256, 0, stream>>>(cur, bf_cur);

        gemm1x1_kernel<0><<<tilesN, 128, 0, stream>>>(wb_q, q_b, bf_cur, qf, 64,
                                                      nullptr, nullptr, nullptr, nullptr);

        conv3x3x3_kernel<0><<<tilesN, 128, 0, stream>>>(wb_o1, off1_b, bf_cur, o1, nullptr, 64);
        conv3x3x3_kernel<1><<<tilesN, 192, 0, stream>>>(wb_o2, nullptr, o1, nullptr, offf, 96);

        sample_attn_kernel<<<(HEADS * NS) / 256, 256, 0, stream>>>(kvf, qf, offf, fusedb);

        float* dst = out + (size_t)(CB + pass * CB) * NS;
        gemm1x1_kernel<1><<<tilesN, 128, 0, stream>>>(wb_out, out_b, fusedb, dst, 64,
                                                      bn_g, bn_bt, bn_mu, bn_v);
    }
}
